// MultiHeadedAttention_73229192397337
// MI455X (gfx1250) — matrix operand order repaired and checked by reading, not compiled, not run on hardware
//
#include <hip/hip_runtime.h>

// ---------------------------------------------------------------------------
// MultiHeadedAttention forward for MI455X (gfx1250, wave32, WMMA)
// B=4, S=2048, D=1024, H=16, DK=64
// f16 WMMA 16x16x32 everywhere, f32 accumulation + f32 online softmax.
// Matrix-core bound: ~137 GFLOP vs ~11us of HBM traffic at 23.3 TB/s.
// ---------------------------------------------------------------------------

typedef _Float16 half_t;
typedef __attribute__((ext_vector_type(16))) _Float16 v16h;
typedef __attribute__((ext_vector_type(8)))  _Float16 v8h;
typedef __attribute__((ext_vector_type(8)))  float    v8f;
typedef __attribute__((ext_vector_type(4)))  int      v4i_t;

#define B_  4
#define S_  2048
#define D_  1024
#define H_  16
#define DK_ 64

// ---- optional CDNA5 async global->LDS path (ASYNCcnt) ----------------------
#if defined(__has_builtin)
#  if __has_builtin(__builtin_amdgcn_global_load_async_to_lds_b128)
#    define HAVE_ASYNC_LDS 1
#  endif
#  if __has_builtin(__builtin_amdgcn_s_wait_asynccnt)
#    define HAVE_WAIT_ASYNC 1
#  endif
#endif
#ifndef HAVE_ASYNC_LDS
#  define HAVE_ASYNC_LDS 0
#endif
#ifndef HAVE_WAIT_ASYNC
#  define HAVE_WAIT_ASYNC 0
#endif

__device__ __forceinline__ void copy16_to_lds(half_t* lds, const half_t* g) {
#if HAVE_ASYNC_LDS
  __builtin_amdgcn_global_load_async_to_lds_b128(
      (__attribute__((address_space(1))) v4i_t*)(void*)g,
      (__attribute__((address_space(3))) v4i_t*)(void*)lds, 0, 0);
#else
  *(v8h*)lds = *(const v8h*)g;
#endif
}

__device__ __forceinline__ void wait_async_copies() {
#if HAVE_ASYNC_LDS
#  if HAVE_WAIT_ASYNC
  __builtin_amdgcn_s_wait_asynccnt(0);
#  else
  asm volatile("s_wait_asynccnt 0" ::: "memory");
#  endif
#endif
}

// ---------------------------------------------------------------------------
__device__ __forceinline__ v8f fzero8() {
  v8f z = {0.f, 0.f, 0.f, 0.f, 0.f, 0.f, 0.f, 0.f};
  return z;
}

__device__ __forceinline__ v8f wmma16x16x32(v16h a, v16h b, v8f c) {
  // D = A(16x32 f16) x B(32x16 f16) + C(16x16 f32)
  return __builtin_amdgcn_wmma_f32_16x16x32_f16(false, a, false, b,
                                                (short)0, c, false, false);
}

// A fragment: 16x32 (MxK) f16. Source row-major: src[row*ld + k].
// lanes 0-15: row M=lane, K halves {0..7,16..23}; lanes 16-31: row M=lane-16,
// K halves {8..15,24..31}.
__device__ __forceinline__ v16h load_a_frag(const half_t* src, int ld) {
  const int lane = threadIdx.x & 31;
  const int row  = lane & 15;
  const int kb   = (lane < 16) ? 0 : 8;
  const half_t* p = src + (size_t)row * ld + kb;
  v8h lo = *(const v8h*)(p);
  v8h hi = *(const v8h*)(p + 16);
  v16h r;
#pragma unroll
  for (int i = 0; i < 8; ++i) { r[i] = lo[i]; r[8 + i] = hi[i]; }
  return r;
}

// B fragment: 32x16 (KxN) f16 where B[k][n] = src[n*ld + k]
// lanes 0-15: N=lane, K {0..7,16..23}; lanes 16-31: N=lane-16, K {8..15,24..31}.
__device__ __forceinline__ v16h load_b_frag(const half_t* src, int ld) {
  const int lane = threadIdx.x & 31;
  const int n    = lane & 15;
  const int kb   = (lane < 16) ? 0 : 8;
  const half_t* p = src + (size_t)n * ld + kb;
  v8h lo = *(const v8h*)(p);
  v8h hi = *(const v8h*)(p + 16);
  v16h r;
#pragma unroll
  for (int i = 0; i < 8; ++i) { r[i] = lo[i]; r[8 + i] = hi[i]; }
  return r;
}

// Row reductions across the 16 lanes holding one C-matrix row (wave32).
__device__ __forceinline__ float rowmax16(float v) {
#pragma unroll
  for (int m = 8; m >= 1; m >>= 1) v = fmaxf(v, __shfl_xor(v, m, 32));
  return v;
}
__device__ __forceinline__ float rowsum16(float v) {
#pragma unroll
  for (int m = 8; m >= 1; m >>= 1) v += __shfl_xor(v, m, 32);
  return v;
}

// ---------------------------------------------------------------------------
// f32 -> f16 conversion (bandwidth trivial)
// ---------------------------------------------------------------------------
__global__ void f32_to_f16_kernel(const float* __restrict__ x,
                                  half_t* __restrict__ y, int n) {
  int i = blockIdx.x * blockDim.x + threadIdx.x;
  int stride = gridDim.x * blockDim.x;
  for (; i < n; i += stride) y[i] = (half_t)x[i];
}

// ---------------------------------------------------------------------------
// Y = X @ W^T + bias    (X: 8192x1024 f16, W: 1024x1024 f16 row-major)
// HEADSPLIT=1: store f16 into (B,H,S,DK) layout (Q/K/V projections)
// HEADSPLIT=0: store f32 row-major (final output projection)
// Block: 256 threads = 8 waves; wave tile 32(M)x64(N) -> 8 WMMA per K-step.
// Block tile 256x64. Grid: (8192/256, 1024/64).
// ---------------------------------------------------------------------------
template <int HEADSPLIT>
__global__ __launch_bounds__(256) void gemm_xwT_kernel(
    const half_t* __restrict__ X, const half_t* __restrict__ W,
    const float* __restrict__ bias, void* __restrict__ Yv) {
  const int wave = threadIdx.x >> 5;
  const int lane = threadIdx.x & 31;
  const int m0 = blockIdx.x * 256 + wave * 32;
  const int n0 = blockIdx.y * 64;

  v8f acc[2][4];
#pragma unroll
  for (int i = 0; i < 2; ++i)
#pragma unroll
    for (int t = 0; t < 4; ++t) acc[i][t] = fzero8();

  const half_t* xp0 = X + (size_t)m0 * D_;
  const half_t* xp1 = X + (size_t)(m0 + 16) * D_;

  for (int kk = 0; kk < D_; kk += 32) {
    if (kk + 32 < D_) {
      __builtin_prefetch(xp0 + kk + 32, 0, 0);                 // global_prefetch_b8
      __builtin_prefetch(xp1 + kk + 32, 0, 0);
      __builtin_prefetch(W + (size_t)(n0 + lane) * D_ + kk + 32, 0, 0);
    }
    v16h a0 = load_a_frag(xp0 + kk, D_);
    v16h a1 = load_a_frag(xp1 + kk, D_);
    v16h b[4];
#pragma unroll
    for (int t = 0; t < 4; ++t)
      b[t] = load_b_frag(W + (size_t)(n0 + 16 * t) * D_ + kk, D_);
#pragma unroll
    for (int t = 0; t < 4; ++t) {
      acc[0][t] = wmma16x16x32(a0, b[t], acc[0][t]);
      acc[1][t] = wmma16x16x32(a1, b[t], acc[1][t]);
    }
  }

  const int row_half = (lane < 16) ? 0 : 8;
  const int col_l = lane & 15;
#pragma unroll
  for (int i = 0; i < 2; ++i) {
#pragma unroll
    for (int t = 0; t < 4; ++t) {
      const int n = n0 + 16 * t + col_l;
      const float bn = bias[n];
#pragma unroll
      for (int r = 0; r < 8; ++r) {
        const int m = m0 + 16 * i + r + row_half;
        const float val = acc[i][t][r] + bn;
        if (HEADSPLIT) {
          const int bb = m >> 11;          // m / S
          const int ss = m & (S_ - 1);
          const int hh = n >> 6;           // n / DK
          const int dk = n & (DK_ - 1);
          ((half_t*)Yv)[((((size_t)bb * H_ + hh) * S_) + ss) * DK_ + dk] =
              (half_t)val;
        } else {
          ((float*)Yv)[(size_t)m * D_ + n] = val;
        }
      }
    }
  }
}

// ---------------------------------------------------------------------------
// Causal flash attention: scores=QK^T/8, causal mask, online softmax, ctx=P@V.
// Block: 128 threads = 4 waves; block handles one (b,h) and 64 q rows;
// each wave owns 16 q rows. Key blocks of 64 (16 WMMA per iteration).
// K tile staged via async global->LDS; V tile staged transposed; P re-laid
// through LDS. B-fragments for each WMMA pass are batched into distinct
// registers so ds_loads clause together and WMMAs issue back-to-back.
// ---------------------------------------------------------------------------
__global__ __launch_bounds__(128) void attn_kernel(
    const half_t* __restrict__ q16, const half_t* __restrict__ k16,
    const half_t* __restrict__ v16, half_t* __restrict__ ctx16) {
  const int bh = blockIdx.y;
  const int b = bh >> 4;      // / H
  const int h = bh & 15;      // % H
  const int wave = threadIdx.x >> 5;
  const int lane = threadIdx.x & 31;
  const int qsuper = blockIdx.x * 64;
  const int qbase = qsuper + wave * 16;

  const half_t* Qb = q16 + (size_t)bh * S_ * DK_;
  const half_t* Kb = k16 + (size_t)bh * S_ * DK_;
  const half_t* Vb = v16 + (size_t)bh * S_ * DK_;

  // Q fragments held in registers for the whole kernel (DK=64 -> 2 K-chunks)
  const v16h qf0 = load_a_frag(Qb + (size_t)qbase * DK_ + 0, DK_);
  const v16h qf1 = load_a_frag(Qb + (size_t)qbase * DK_ + 32, DK_);

  __shared__ __align__(16) half_t Kl[64][72];     // K tile row-major [key][dk]
  __shared__ __align__(16) half_t Vt[64][72];     // V^T tile [dk][key]
  __shared__ __align__(16) half_t Pl[4][16][72];  // per-wave P tile (16x64)

  float m_r[8], l_r[8];
  v8f acc[4] = {fzero8(), fzero8(), fzero8(), fzero8()};
#pragma unroll
  for (int r = 0; r < 8; ++r) { m_r[r] = -3.0e38f; l_r[r] = 0.f; }

  const int row_half = (lane < 16) ? 0 : 8;
  const int col_l = lane & 15;
  const int kend = qsuper + 64;   // causal: no key block beyond q_max

  for (int k0 = 0; k0 < kend; k0 += 64) {
    __syncthreads();   // all waves done reading previous K/V tiles

    // ---- stage K tile (64 keys x 64 dk) via async global->LDS copies ----
    {
      const int key = threadIdx.x >> 1;          // 0..63
      const int seg = (threadIdx.x & 1) * 32;    // dk 0 or 32
      const half_t* gp = Kb + (size_t)(k0 + key) * DK_ + seg;
      half_t* lp = &Kl[key][seg];
#pragma unroll
      for (int j = 0; j < 4; ++j) copy16_to_lds(lp + 8 * j, gp + 8 * j);
    }
    // ---- stage V^T tile (transpose at store) ----
    {
      const int key = threadIdx.x & 63;          // 0..63
      const int seg = (threadIdx.x >> 6) * 32;   // dk 0 or 32
      const half_t* vp = Vb + (size_t)(k0 + key) * DK_ + seg;
      v8h v0 = *(const v8h*)(vp);
      v8h v1 = *(const v8h*)(vp + 8);
      v8h v2 = *(const v8h*)(vp + 16);
      v8h v3 = *(const v8h*)(vp + 24);
#pragma unroll
      for (int i = 0; i < 8; ++i) {
        Vt[seg + i][key]      = v0[i];
        Vt[seg + 8 + i][key]  = v1[i];
        Vt[seg + 16 + i][key] = v2[i];
        Vt[seg + 24 + i][key] = v3[i];
      }
    }
    wait_async_copies();
    __syncthreads();

    // ---- scores: four 16x16 tiles covering keys [k0, k0+64) ----
    // Batch all 8 B-fragments first so the ds_loads clause together and the
    // 8 WMMAs issue as a burst (XDL co-executes with trailing VALU).
    v16h kbf[8];
#pragma unroll
    for (int kt = 0; kt < 4; ++kt) {
      kbf[2 * kt]     = load_b_frag(&Kl[kt * 16][0], 72);
      kbf[2 * kt + 1] = load_b_frag(&Kl[kt * 16][32], 72);
    }
    v8f s[4];
#pragma unroll
    for (int kt = 0; kt < 4; ++kt) {
      v8f sc = fzero8();
      sc = wmma16x16x32(qf0, kbf[2 * kt], sc);
      sc = wmma16x16x32(qf1, kbf[2 * kt + 1], sc);
      s[kt] = sc;
    }

    // ---- scale, causal mask, online softmax update ----
#pragma unroll
    for (int r = 0; r < 8; ++r) {
      const int rowq = qbase + r + row_half;
      float sv[4];
#pragma unroll
      for (int kt = 0; kt < 4; ++kt) {
        float x = s[kt][r] * 0.125f;            // 1/sqrt(DK)
        if (k0 + kt * 16 + col_l > rowq) x = -1.0e9f;
        sv[kt] = x;
      }
      float mx = fmaxf(fmaxf(sv[0], sv[1]), fmaxf(sv[2], sv[3]));
      mx = rowmax16(mx);
      const float nm = fmaxf(m_r[r], mx);
      const float alpha = __expf(m_r[r] - nm);
      float p[4];
#pragma unroll
      for (int kt = 0; kt < 4; ++kt) p[kt] = __expf(sv[kt] - nm);
      const float rs = rowsum16((p[0] + p[1]) + (p[2] + p[3]));
      l_r[r] = l_r[r] * alpha + rs;
      m_r[r] = nm;
#pragma unroll
      for (int t = 0; t < 4; ++t) acc[t][r] *= alpha;
      const int rr = r + row_half;
#pragma unroll
      for (int kt = 0; kt < 4; ++kt)
        Pl[wave][rr][kt * 16 + col_l] = (half_t)p[kt];
    }
    // within-wave LDS write -> read fence (DS ops tracked by DScnt)
    asm volatile("s_wait_dscnt 0" ::: "memory");

    // ---- ctx += P(16x64) @ V(64x64): 8 WMMAs, B-frags batched ----
    const v16h pf0 = load_a_frag(&Pl[wave][0][0], 72);
    const v16h pf1 = load_a_frag(&Pl[wave][0][32], 72);
    v16h vbf[8];
#pragma unroll
    for (int t = 0; t < 4; ++t) {
      vbf[2 * t]     = load_b_frag(&Vt[t * 16][0], 72);
      vbf[2 * t + 1] = load_b_frag(&Vt[t * 16][32], 72);
    }
#pragma unroll
    for (int t = 0; t < 4; ++t) {
      v8f a = acc[t];
      a = wmma16x16x32(pf0, vbf[2 * t], a);
      a = wmma16x16x32(pf1, vbf[2 * t + 1], a);
      acc[t] = a;
    }
  }

  // ---- normalize, store ctx f16 row-major (b*S+s, D), col = h*64+dk ----
  half_t* Ob = ctx16 + (size_t)b * S_ * D_ + h * DK_;
#pragma unroll
  for (int r = 0; r < 8; ++r) {
    const float inv = 1.0f / l_r[r];
    const int rowq = qbase + r + row_half;
#pragma unroll
    for (int t = 0; t < 4; ++t) {
      Ob[(size_t)rowq * D_ + t * 16 + col_l] = (half_t)(acc[t][r] * inv);
    }
  }
}

// ---------------------------------------------------------------------------
extern "C" void kernel_launch(void* const* d_in, const int* in_sizes, int n_in,
                              void* d_out, int out_size, void* d_ws,
                              size_t ws_size, hipStream_t stream) {
  const float* query = (const float*)d_in[0];
  const float* key_  = (const float*)d_in[1];
  const float* value = (const float*)d_in[2];
  // d_in[3] is the tril mask; causal masking is applied analytically.
  const float* Wq = (const float*)d_in[4];
  const float* bq = (const float*)d_in[5];
  const float* Wk = (const float*)d_in[6];
  const float* bk = (const float*)d_in[7];
  const float* Wv = (const float*)d_in[8];
  const float* bv = (const float*)d_in[9];
  const float* Wo = (const float*)d_in[10];
  const float* bo = (const float*)d_in[11];
  float* out = (float*)d_out;

  const size_t BSD = (size_t)B_ * S_ * D_;   // 8,388,608
  const size_t DD  = (size_t)D_ * D_;        // 1,048,576

  char* w = (char*)d_ws;
  half_t* xq16 = (half_t*)w; w += BSD * 2;
  half_t* xk16 = (half_t*)w; w += BSD * 2;
  half_t* xv16 = (half_t*)w; w += BSD * 2;
  half_t* Wq16 = (half_t*)w; w += DD * 2;
  half_t* Wk16 = (half_t*)w; w += DD * 2;
  half_t* Wv16 = (half_t*)w; w += DD * 2;
  half_t* Wo16 = (half_t*)w; w += DD * 2;
  half_t* q16  = (half_t*)w; w += BSD * 2;
  half_t* k16  = (half_t*)w; w += BSD * 2;
  half_t* v16  = (half_t*)w; w += BSD * 2;
  half_t* ctx16 = xq16;   // reuse: xq16 is dead after the Q projection

  dim3 cblk(256), cgrd(1024);
  f32_to_f16_kernel<<<cgrd, cblk, 0, stream>>>(query, xq16, (int)BSD);
  f32_to_f16_kernel<<<cgrd, cblk, 0, stream>>>(key_,  xk16, (int)BSD);
  f32_to_f16_kernel<<<cgrd, cblk, 0, stream>>>(value, xv16, (int)BSD);
  f32_to_f16_kernel<<<cgrd, cblk, 0, stream>>>(Wq, Wq16, (int)DD);
  f32_to_f16_kernel<<<cgrd, cblk, 0, stream>>>(Wk, Wk16, (int)DD);
  f32_to_f16_kernel<<<cgrd, cblk, 0, stream>>>(Wv, Wv16, (int)DD);
  f32_to_f16_kernel<<<cgrd, cblk, 0, stream>>>(Wo, Wo16, (int)DD);

  // Projections: M=B*S=8192, N=K=D=1024. Grid: (8192/256, 1024/64)
  dim3 gblk(256), ggrd(32, 16);
  gemm_xwT_kernel<1><<<ggrd, gblk, 0, stream>>>(xq16, Wq16, bq, (void*)q16);
  gemm_xwT_kernel<1><<<ggrd, gblk, 0, stream>>>(xk16, Wk16, bk, (void*)k16);
  gemm_xwT_kernel<1><<<ggrd, gblk, 0, stream>>>(xv16, Wv16, bv, (void*)v16);

  // Attention: grid (S/64, B*H)
  dim3 ablk(128), agrd(S_ / 64, B_ * H_);
  attn_kernel<<<agrd, ablk, 0, stream>>>(q16, k16, v16, ctx16);

  // Output projection (f32 out)
  gemm_xwT_kernel<0><<<ggrd, gblk, 0, stream>>>(ctx16, Wo16, bo, (void*)out);
}
